// OverlapGATNet_59399397704031
// MI455X (gfx1250) — compile-verified, matrix operand order checked by reading, not run on hardware
//
#include <hip/hip_runtime.h>
#include <math.h>

typedef __attribute__((ext_vector_type(16))) _Float16 v16h;
typedef __attribute__((ext_vector_type(8)))  _Float16 v8h;
typedef __attribute__((ext_vector_type(8)))  float    v8f;

#define NB    2        // batch
#define GH    8
#define GW    112
#define NN    896      // GH*GW nodes
#define TK1   500
#define TK2   300
#define RADI  9
#define WSIDE 19
#define NEDGE 361      // 19*19

// ---------------------------------------------------------------------------
// generic helpers
// ---------------------------------------------------------------------------
__global__ void cvt_f16_kernel(const float* __restrict__ s, _Float16* __restrict__ d, int n) {
  int i = blockIdx.x * blockDim.x + threadIdx.x;
  if (i < n) d[i] = (_Float16)s[i];
}

__global__ void zero_kernel(float* __restrict__ p, int n) {
  int i = blockIdx.x * blockDim.x + threadIdx.x;
  if (i < n) p[i] = 0.f;
}

// load one 16-half WMMA fragment as two 16-byte vector loads
// (lane's halves are exactly runs [kb..kb+7] and [16+kb..16+kb+7])
__device__ __forceinline__ v16h frag_ld(const _Float16* __restrict__ p, int kb) {
  v8h lo = *(const v8h*)(p + kb);
  v8h hi = *(const v8h*)(p + 16 + kb);
  return __builtin_shufflevector(lo, hi, 0, 1, 2, 3, 4, 5, 6, 7, 8, 9, 10, 11, 12, 13, 14, 15);
}

// ---------------------------------------------------------------------------
// implicit-GEMM conv via WMMA f16 (A = weights [Cout,K], B = input patches [K,HW])
// one wave per 16x16 output tile
// ---------------------------------------------------------------------------
__device__ __forceinline__ v16h conv_bfrag(const _Float16* __restrict__ inb,
                                           int k0, int kb, int K, int KsKs, int Ks,
                                           int oy, int ox, int stride, int pad,
                                           int Hin, int Win) {
  v16h bf;
#pragma unroll
  for (int i = 0; i < 16; ++i) {
    const int ko = (i < 8) ? (kb + i) : (16 + kb + (i - 8));
    const int k  = k0 + ko;
    _Float16 bv = (_Float16)0.f;
    if (k < K) {
      const int ci = k / KsKs;
      const int r  = k - ci * KsKs;
      const int ky = r / Ks, kx = r - (r / Ks) * Ks;
      const int iy = oy * stride + ky - pad;
      const int ix = ox * stride + kx - pad;
      if (iy >= 0 && iy < Hin && ix >= 0 && ix < Win)
        bv = inb[((size_t)ci * Hin + iy) * Win + ix];
    }
    bf[i] = bv;
  }
  return bf;
}

__global__ __launch_bounds__(32) void conv_wmma_kernel(
    const _Float16* __restrict__ in, const _Float16* __restrict__ w,
    float* __restrict__ out,
    int Cin, int Hin, int Win, int Cout, int Ks, int stride, int pad,
    int Hout, int Wout) {
  const int lane = threadIdx.x & 31;
  const int b = blockIdx.z;
  const int HW = Hout * Wout;
  const int K  = Cin * Ks * Ks;
  const int KsKs = Ks * Ks;
  const int m0 = blockIdx.y * 16;
  const int n0 = blockIdx.x * 16;
  const int ln = lane & 15;
  const int kb = (lane & 16) ? 8 : 0;     // K-half per lane group (16-bit A/B layout)
  const int n  = n0 + ln;
  const int oy = n / Wout, ox = n - (n / Wout) * Wout;
  const _Float16* __restrict__ inb  = in + (size_t)b * Cin * Hin * Win;
  const _Float16* __restrict__ wrow = w + (size_t)(m0 + ln) * K;
  v8f c = {};
  if ((K & 31) == 0) {
    // vectorized A-fragment path (all layers except conv1)
    for (int k0 = 0; k0 < K; k0 += 32) {
      v16h a  = frag_ld(wrow + k0, kb);
      v16h bf = conv_bfrag(inb, k0, kb, K, KsKs, Ks, oy, ox, stride, pad, Hin, Win);
      c = __builtin_amdgcn_wmma_f32_16x16x32_f16(false, a, false, bf, (short)0, c, false, false);
    }
  } else {
    // guarded scalar A path (K not a multiple of 32, e.g. conv1 K=25)
    for (int k0 = 0; k0 < K; k0 += 32) {
      v16h a;
#pragma unroll
      for (int i = 0; i < 16; ++i) {
        const int ko = (i < 8) ? (kb + i) : (16 + kb + (i - 8));
        const int k  = k0 + ko;
        a[i] = (k < K) ? wrow[k] : (_Float16)0.f;
      }
      v16h bf = conv_bfrag(inb, k0, kb, K, KsKs, Ks, oy, ox, stride, pad, Hin, Win);
      c = __builtin_amdgcn_wmma_f32_16x16x32_f16(false, a, false, bf, (short)0, c, false, false);
    }
  }
  const int mrow0 = m0 + ((lane >> 4) ? 8 : 0);
  float* op = out + ((size_t)b * Cout + mrow0) * HW + n;
#pragma unroll
  for (int r = 0; r < 8; ++r) op[(size_t)r * HW] = c[r];
}

// ---------------------------------------------------------------------------
// GroupNorm: stats (mean, rstd) then normalize+ReLU (+f16 cast, optional f32 out)
// ---------------------------------------------------------------------------
__global__ void gn_stats_kernel(const float* __restrict__ x, float* __restrict__ stats,
                                int C, int HW, int groups) {
  const int bg = blockIdx.x;
  const int b = bg / groups, g = bg - (bg / groups) * groups;
  const int cpg = C / groups;
  const size_t cnt = (size_t)cpg * HW;
  const float* p = x + ((size_t)b * C + (size_t)g * cpg) * HW;
  float s = 0.f, ss = 0.f;
  for (size_t i = threadIdx.x; i < cnt; i += blockDim.x) { float v = p[i]; s += v; ss += v * v; }
  __shared__ float rs[1024], rss[1024];
  rs[threadIdx.x] = s; rss[threadIdx.x] = ss; __syncthreads();
  for (int o = blockDim.x >> 1; o > 0; o >>= 1) {
    if ((int)threadIdx.x < o) { rs[threadIdx.x] += rs[threadIdx.x + o]; rss[threadIdx.x] += rss[threadIdx.x + o]; }
    __syncthreads();
  }
  if (threadIdx.x == 0) {
    float m = rs[0] / (float)cnt;
    float var = rss[0] / (float)cnt - m * m;
    stats[bg * 2] = m;
    stats[bg * 2 + 1] = rsqrtf(var + 1e-5f);
  }
}

__global__ void gn_norm_kernel(const float* __restrict__ x, const float* __restrict__ stats,
                               const float* __restrict__ gw, const float* __restrict__ gb,
                               _Float16* __restrict__ out16, float* __restrict__ out32,
                               int C, int HW, int groups) {
  const size_t idx = (size_t)blockIdx.x * blockDim.x + threadIdx.x;
  const size_t total = (size_t)NB * C * HW;
  if (idx >= total) return;
  const int c = (int)((idx / HW) % C);
  const int b = (int)(idx / ((size_t)C * HW));
  const int g = c / (C / groups);
  const float m = stats[(b * groups + g) * 2];
  const float r = stats[(b * groups + g) * 2 + 1];
  float v = (x[idx] - m) * r * gw[c] + gb[c];
  v = v > 0.f ? v : 0.f;
  out16[idx] = (_Float16)v;
  if (out32) out32[idx] = v;
}

// ---------------------------------------------------------------------------
// pool [B,128,16,225] -> pf [B,896,128] (+f16 copy): mean over H-pair + adaptive W bins
// ---------------------------------------------------------------------------
__global__ void pool_kernel(const float* __restrict__ x, float* __restrict__ pf,
                            _Float16* __restrict__ pf16) {
  const int node = blockIdx.x;      // 0..895
  const int b = blockIdx.y;
  const int c = threadIdx.x;        // 128
  const int gy = node / GW, gx = node - (node / GW) * GW;
  const int ws = (gx * 225) / GW;
  const int we = ((gx + 1) * 225 + GW - 1) / GW;
  float s = 0.f;
  for (int h = 2 * gy; h < 2 * gy + 2; ++h)
    for (int w = ws; w < we; ++w)
      s += x[(((size_t)b * 128 + c) * 16 + h) * 225 + w];
  s /= (float)(2 * (we - ws));
  const size_t o = ((size_t)b * NN + node) * 128 + c;
  pf[o] = s;
  pf16[o] = (_Float16)s;
}

// ---------------------------------------------------------------------------
// row L2 norms (scores)
// ---------------------------------------------------------------------------
__global__ void row_norm_kernel(const float* __restrict__ x, float* __restrict__ out, int dim) {
  const int row = blockIdx.x;
  float s = 0.f;
  for (int i = threadIdx.x; i < dim; i += blockDim.x) { float v = x[(size_t)row * dim + i]; s += v * v; }
  __shared__ float red[256];
  red[threadIdx.x] = s; __syncthreads();
  for (int o = 128; o > 0; o >>= 1) { if ((int)threadIdx.x < o) red[threadIdx.x] += red[threadIdx.x + o]; __syncthreads(); }
  if (threadIdx.x == 0) out[row] = sqrtf(red[0]);
}

// ---------------------------------------------------------------------------
// top-k via one-block bitonic sort (order-invariant downstream -> set selection OK)
// ---------------------------------------------------------------------------
__global__ void topk_kernel(const float* __restrict__ score, const int* __restrict__ mask_in,
                            int n, int k, int* __restrict__ mask_out, int* __restrict__ list_out) {
  const int b = blockIdx.x;
  const int tid = threadIdx.x;     // 512
  __shared__ float sv[1024];
  __shared__ int   si[1024];
  for (int i = tid; i < 1024; i += 512) {
    float v = -1e30f;
    if (i < n) {
      v = score[b * n + i];
      if (mask_in && !mask_in[b * n + i]) v = -1e30f;
    }
    sv[i] = v; si[i] = i;
  }
  __syncthreads();
  for (int kk = 2; kk <= 1024; kk <<= 1) {
    for (int j = kk >> 1; j > 0; j >>= 1) {
      for (int i = tid; i < 1024; i += 512) {
        const int ixj = i ^ j;
        if (ixj > i) {
          const bool desc = ((i & kk) == 0);
          const bool sw = desc ? (sv[i] < sv[ixj]) : (sv[i] > sv[ixj]);
          if (sw) {
            float tv = sv[i]; sv[i] = sv[ixj]; sv[ixj] = tv;
            int ti = si[i]; si[i] = si[ixj]; si[ixj] = ti;
          }
        }
      }
      __syncthreads();
    }
  }
  if (mask_out) {
    for (int i = tid; i < n; i += 512) mask_out[b * n + i] = 0;
    __syncthreads();
    for (int i = tid; i < k; i += 512) mask_out[b * n + si[i]] = 1;
  }
  if (list_out)
    for (int i = tid; i < k; i += 512) list_out[b * k + i] = si[i];
}

// ---------------------------------------------------------------------------
// linear projection via WMMA: out[rows,256] = A[rows,K] * W[256,K]^T + bias
// K in {128,256}; vectorized b128 fragment loads, K-loop unrolled x2
// ---------------------------------------------------------------------------
__global__ __launch_bounds__(32) void gemm_lin_kernel(
    const _Float16* __restrict__ A, const _Float16* __restrict__ W,
    const float* __restrict__ bias, float* __restrict__ out, int K) {
  const int lane = threadIdx.x & 31;
  const int r0 = blockIdx.x * 16;
  const int n0 = blockIdx.y * 16;
  const int ln = lane & 15;
  const int kb = (lane & 16) ? 8 : 0;
  const _Float16* ap = A + (size_t)(r0 + ln) * K;
  const _Float16* bp = W + (size_t)(n0 + ln) * K;
  v8f c = {};
  for (int k0 = 0; k0 < K; k0 += 64) {
    v16h a0 = frag_ld(ap + k0, kb);
    v16h b0 = frag_ld(bp + k0, kb);
    v16h a1 = frag_ld(ap + k0 + 32, kb);
    v16h b1 = frag_ld(bp + k0 + 32, kb);
    c = __builtin_amdgcn_wmma_f32_16x16x32_f16(false, a0, false, b0, (short)0, c, false, false);
    c = __builtin_amdgcn_wmma_f32_16x16x32_f16(false, a1, false, b1, (short)0, c, false, false);
  }
  const int n = n0 + ln;
  const int r = r0 + ((lane >> 4) ? 8 : 0);
  const float bv = bias[n];
#pragma unroll
  for (int i = 0; i < 8; ++i) out[(size_t)(r + i) * 256 + n] = c[i] + bv;
}

// ---------------------------------------------------------------------------
// GATv2 per-destination softmax aggregation (reverse window + top-k mask)
// ---------------------------------------------------------------------------
__global__ void gat_agg_kernel(const float* __restrict__ xl, const float* __restrict__ xr,
                               const float* __restrict__ att, const int* __restrict__ mask,
                               float* __restrict__ agg) {
  const int d = blockIdx.x;     // dst node 0..895
  const int b = blockIdx.y;
  const int tid = threadIdx.x;  // 256
  __shared__ float xr_s[256], att_s[256];
  __shared__ float ev[NEDGE];
  __shared__ int   es[NEDGE];
  __shared__ float red[256];
  const int dy = d / GW, dx = d - (d / GW) * GW;
  xr_s[tid] = xr[((size_t)(b * NN + d)) * 256 + tid];
  att_s[tid] = att[tid];
  for (int t = tid; t < NEDGE; t += 256) {
    const int ry = t / WSIDE - RADI, rx = t - (t / WSIDE) * WSIDE - RADI;
    const int sy = dy - ry, sx = dx - rx;
    const int s = sy * GW + sx;
    const bool ok = (sy >= 0 && sy < GH && sx >= 0 && sx < GW) && (mask[b * NN + s] != 0);
    es[t] = ok ? s : -1;
    ev[t] = -1e30f;
  }
  __syncthreads();
  const int wave = tid >> 5, lane = tid & 31;
  for (int t = wave; t < NEDGE; t += 8) {
    const int s = es[t];
    if (s < 0) continue;               // uniform across the wave
    const float* xls = xl + ((size_t)(b * NN + s)) * 256;
    float acc = 0.f;
#pragma unroll
    for (int j = 0; j < 8; ++j) {
      const int f = lane + 32 * j;
      float m = xls[f] + xr_s[f];
      m = m > 0.f ? m : 0.2f * m;      // leaky_relu(0.2)
      acc += m * att_s[f];
    }
    for (int o = 16; o > 0; o >>= 1) acc += __shfl_xor(acc, o, 32);
    if (lane == 0) ev[t] = acc;
  }
  __syncthreads();
  float mx = -1e30f;
  for (int t = tid; t < NEDGE; t += 256) mx = fmaxf(mx, ev[t]);
  red[tid] = mx; __syncthreads();
  for (int o = 128; o > 0; o >>= 1) { if (tid < o) red[tid] = fmaxf(red[tid], red[tid + o]); __syncthreads(); }
  const float emax = red[0];
  __syncthreads();
  float* aggrow = agg + ((size_t)(b * NN + d)) * 256;
  if (emax <= -1e29f) { aggrow[tid] = 0.f; return; }     // no incoming edges
  for (int t = tid; t < NEDGE; t += 256)
    ev[t] = (es[t] >= 0) ? __expf(ev[t] - emax) : 0.f;
  __syncthreads();
  float sm = 0.f;
  for (int t = tid; t < NEDGE; t += 256) sm += ev[t];
  red[tid] = sm; __syncthreads();
  for (int o = 128; o > 0; o >>= 1) { if (tid < o) red[tid] += red[tid + o]; __syncthreads(); }
  const float denom = fmaxf(red[0], 1e-16f);
  float acc = 0.f;
  for (int t = 0; t < NEDGE; ++t) {
    const int s = es[t];
    if (s >= 0) acc += ev[t] * xl[((size_t)(b * NN + s)) * 256 + tid];
  }
  aggrow[tid] = acc / denom;
}

__global__ void gat_combine_kernel(const float* __restrict__ agg, const float* __restrict__ res,
                                   float* __restrict__ h, _Float16* __restrict__ h16, int n) {
  const int i = blockIdx.x * blockDim.x + threadIdx.x;
  if (i >= n) return;
  float v = agg[i] + res[i];
  v = v > 0.f ? v : 0.f;
  h[i] = v;
  if (h16) h16[i] = (_Float16)v;
}

// ---------------------------------------------------------------------------
// gather selected 300 rows and L2-normalize (eps 1e-12)
// ---------------------------------------------------------------------------
__global__ void gather_norm_kernel(const float* __restrict__ h2, const int* __restrict__ sel,
                                   float* __restrict__ g) {
  const int j = blockIdx.x, b = blockIdx.y, tid = threadIdx.x;  // 256
  const int node = sel[b * TK2 + j];
  const float v = h2[((size_t)(b * NN + node)) * 256 + tid];
  __shared__ float red[256];
  red[tid] = v * v; __syncthreads();
  for (int o = 128; o > 0; o >>= 1) { if (tid < o) red[tid] += red[tid + o]; __syncthreads(); }
  const float nrm = fmaxf(sqrtf(red[0]), 1e-12f);
  g[((size_t)(b * TK2 + j)) * 256 + tid] = v / nrm;
}

// ---------------------------------------------------------------------------
// NetVLAD
// ---------------------------------------------------------------------------
__global__ void nv_act_kernel(const float* __restrict__ g, const float* __restrict__ wc,
                              const float* __restrict__ bc, float* __restrict__ act) {
  const int n = blockIdx.x, b = blockIdx.y, k = threadIdx.x;   // 64
  __shared__ float row[256];
  __shared__ float red[64];
  for (int i = k; i < 256; i += 64) row[i] = g[((size_t)(b * TK2 + n)) * 256 + i];
  __syncthreads();
  float acc = bc[k];
  for (int f = 0; f < 256; ++f) acc += row[f] * wc[f * 64 + k];
  red[k] = acc; __syncthreads();
  for (int o = 32; o > 0; o >>= 1) { if (k < o) red[k] = fmaxf(red[k], red[k + o]); __syncthreads(); }
  const float mx = red[0]; __syncthreads();
  const float e = __expf(acc - mx);
  red[k] = e; __syncthreads();
  for (int o = 32; o > 0; o >>= 1) { if (k < o) red[k] += red[k + o]; __syncthreads(); }
  act[((size_t)(b * TK2 + n)) * 64 + k] = e / red[0];
}

__global__ void nv_vlad_kernel(const float* __restrict__ act, const float* __restrict__ g,
                               const float* __restrict__ w2, float* __restrict__ vlad) {
  const int k = blockIdx.x, b = blockIdx.y, f = threadIdx.x;   // 256
  float acc = 0.f, s = 0.f;
  for (int n = 0; n < TK2; ++n) {
    const float a = act[((size_t)(b * TK2 + n)) * 64 + k];
    s += a;
    acc += a * g[((size_t)(b * TK2 + n)) * 256 + f];
  }
  vlad[((size_t)(b * 256 + f)) * 64 + k] = acc - s * w2[f * 64 + k];
}

__global__ void nv_intra_kernel(float* __restrict__ vlad) {
  const int k = blockIdx.x, b = blockIdx.y, f = threadIdx.x;   // 256
  const float v = vlad[((size_t)(b * 256 + f)) * 64 + k];
  __shared__ float red[256];
  red[f] = v * v; __syncthreads();
  for (int o = 128; o > 0; o >>= 1) { if (f < o) red[f] += red[f + o]; __syncthreads(); }
  const float nrm = fmaxf(sqrtf(red[0]), 1e-12f);
  vlad[((size_t)(b * 256 + f)) * 64 + k] = v / nrm;
}

__global__ void vec_norm_kernel(const float* __restrict__ x, float* __restrict__ norms, int dim) {
  const int b = blockIdx.x;
  float s = 0.f;
  for (int i = threadIdx.x; i < dim; i += 256) { float v = x[(size_t)b * dim + i]; s += v * v; }
  __shared__ float red[256];
  red[threadIdx.x] = s; __syncthreads();
  for (int o = 128; o > 0; o >>= 1) { if ((int)threadIdx.x < o) red[threadIdx.x] += red[threadIdx.x + o]; __syncthreads(); }
  if (threadIdx.x == 0) norms[b] = fmaxf(sqrtf(red[0]), 1e-12f);
}

// v[b,o] += sum_i (vlad[b,i]/norm[b]) * wh[i,o], i-chunked with atomics
__global__ void nv_hidden_kernel(const float* __restrict__ vlad, const float* __restrict__ norms,
                                 const float* __restrict__ wh, float* __restrict__ v) {
  const int chunk = blockIdx.x;   // 64 chunks of 256 i's
  const int b = blockIdx.y;
  const int o = threadIdx.x;      // 256
  const float rn = 1.f / norms[b];
  float acc = 0.f;
  const int i0 = chunk * 256;
  for (int i = i0; i < i0 + 256; ++i)
    acc += vlad[(size_t)b * 16384 + i] * rn * wh[(size_t)i * 256 + o];
  atomicAdd(&v[b * 256 + o], acc);
}

__global__ void nv_final_kernel(const float* __restrict__ v, const float* __restrict__ bng,
                                const float* __restrict__ bnb, const float* __restrict__ wg,
                                const float* __restrict__ bg, float* __restrict__ out) {
  const int o = threadIdx.x;      // 256
  __shared__ float vbn[2][256];
  __shared__ float ov[2][256];
  __shared__ float red[256];
  const float v0 = v[o], v1 = v[256 + o];
  const float m = 0.5f * (v0 + v1);
  const float var = 0.5f * ((v0 - m) * (v0 - m) + (v1 - m) * (v1 - m));
  const float rs = rsqrtf(var + 1e-5f);
  vbn[0][o] = (v0 - m) * rs * bng[o] + bnb[o];
  vbn[1][o] = (v1 - m) * rs * bng[o] + bnb[o];
  __syncthreads();
  for (int b = 0; b < 2; ++b) {
    float acc = bg[o];
    for (int j = 0; j < 256; ++j) acc += vbn[b][j] * wg[j * 256 + o];
    const float gt = 1.f / (1.f + __expf(-acc));
    ov[b][o] = vbn[b][o] * gt;
  }
  __syncthreads();
  for (int b = 0; b < 2; ++b) {
    red[o] = ov[b][o] * ov[b][o]; __syncthreads();
    for (int s = 128; s > 0; s >>= 1) { if (o < s) red[o] += red[o + s]; __syncthreads(); }
    const float nrm = fmaxf(sqrtf(red[0]), 1e-12f);
    __syncthreads();
    out[b * 256 + o] = ov[b][o] / nrm;
  }
}

// ---------------------------------------------------------------------------
// host driver
// ---------------------------------------------------------------------------
extern "C" void kernel_launch(void* const* d_in, const int* in_sizes, int n_in,
                              void* d_out, int out_size, void* d_ws, size_t ws_size,
                              hipStream_t stream) {
  (void)in_sizes; (void)n_in; (void)out_size; (void)ws_size;
  const float* x = (const float*)d_in[0];
  const float* conv_w[6] = {(const float*)d_in[1], (const float*)d_in[2], (const float*)d_in[3],
                            (const float*)d_in[4], (const float*)d_in[5], (const float*)d_in[6]};
  const float* gnw[6] = {(const float*)d_in[7],  (const float*)d_in[9],  (const float*)d_in[11],
                         (const float*)d_in[13], (const float*)d_in[15], (const float*)d_in[17]};
  const float* gnb[6] = {(const float*)d_in[8],  (const float*)d_in[10], (const float*)d_in[12],
                         (const float*)d_in[14], (const float*)d_in[16], (const float*)d_in[18]};
  const float* g1_lw = (const float*)d_in[19]; const float* g1_lb = (const float*)d_in[20];
  const float* g1_rw = (const float*)d_in[21]; const float* g1_rb = (const float*)d_in[22];
  const float* g1_at = (const float*)d_in[23]; const float* g1_rs = (const float*)d_in[24];
  const float* g1_bi = (const float*)d_in[25];
  const float* g2_lw = (const float*)d_in[26]; const float* g2_lb = (const float*)d_in[27];
  const float* g2_rw = (const float*)d_in[28]; const float* g2_rb = (const float*)d_in[29];
  const float* g2_at = (const float*)d_in[30]; const float* g2_rs = (const float*)d_in[31];
  const float* g2_bi = (const float*)d_in[32];
  const float* nv_cw  = (const float*)d_in[33]; const float* nv_cb  = (const float*)d_in[34];
  const float* nv_cw2 = (const float*)d_in[35]; const float* nv_hw  = (const float*)d_in[36];
  const float* nv_bg  = (const float*)d_in[37]; const float* nv_bb  = (const float*)d_in[38];
  const float* nv_gw  = (const float*)d_in[39]; const float* nv_gbp = (const float*)d_in[40];
  float* outp = (float*)d_out;

  char* wsp = (char*)d_ws;
  size_t off = 0;
  auto carve = [&](size_t bytes) -> void* {
    void* p = wsp + off;
    off = (off + bytes + 255) & ~(size_t)255;
    return p;
  };

  // conv specs: Cin,Hin,Win,Cout,Ks,stride,pad,Hout,Wout,groups
  const int cs[6][10] = {
    {1,   64, 900, 32,  5, 1, 2, 64, 900, 4},
    {32,  64, 900, 32,  3, 1, 1, 64, 900, 4},
    {32,  64, 900, 64,  5, 2, 2, 32, 450, 8},
    {64,  32, 450, 64,  3, 1, 1, 32, 450, 8},
    {64,  32, 450, 128, 5, 2, 2, 16, 225, 16},
    {128, 16, 225, 128, 3, 1, 1, 16, 225, 16}};
  const int wcnt[6] = {800, 9216, 51200, 36864, 204800, 147456};

  _Float16* x16 = (_Float16*)carve((size_t)NB * 64 * 900 * 2);
  _Float16* w16[6];
  for (int l = 0; l < 6; ++l) w16[l] = (_Float16*)carve((size_t)wcnt[l] * 2);
  _Float16* gw16[6]; // g1 l,r,res ; g2 l,r,res
  const int gwcnt[6] = {32768, 32768, 32768, 65536, 65536, 65536};
  const float* gwsrc[6] = {g1_lw, g1_rw, g1_rs, g2_lw, g2_rw, g2_rs};
  for (int l = 0; l < 6; ++l) gw16[l] = (_Float16*)carve((size_t)gwcnt[l] * 2);

  float*    c32    = (float*)carve((size_t)NB * 32 * 64 * 900 * 4);   // max conv out
  float*    g32l   = (float*)carve((size_t)NB * 128 * 16 * 225 * 4);  // last GN f32 out
  _Float16* h16a   = (_Float16*)carve((size_t)NB * 32 * 64 * 900 * 2);
  _Float16* h16b   = (_Float16*)carve((size_t)NB * 32 * 64 * 900 * 2);
  float*    stats  = (float*)carve(2 * 16 * 2 * 4 * NB);
  float*    pf     = (float*)carve((size_t)NB * NN * 128 * 4);
  _Float16* pf16   = (_Float16*)carve((size_t)NB * NN * 128 * 2);
  float*    s1     = (float*)carve((size_t)NB * NN * 4);
  float*    s2f    = (float*)carve((size_t)NB * NN * 4);
  int*      maskb  = (int*)carve((size_t)NB * NN * 4);
  int*      sel    = (int*)carve((size_t)NB * TK2 * 4);
  float*    xl     = (float*)carve((size_t)NB * NN * 256 * 4);
  float*    xr     = (float*)carve((size_t)NB * NN * 256 * 4);
  float*    res    = (float*)carve((size_t)NB * NN * 256 * 4);
  float*    agg    = (float*)carve((size_t)NB * NN * 256 * 4);
  float*    h1     = (float*)carve((size_t)NB * NN * 256 * 4);
  _Float16* h1_16  = (_Float16*)carve((size_t)NB * NN * 256 * 2);
  float*    h2     = (float*)carve((size_t)NB * NN * 256 * 4);
  float*    gsel   = (float*)carve((size_t)NB * TK2 * 256 * 4);
  float*    actn   = (float*)carve((size_t)NB * TK2 * 64 * 4);
  float*    vlad   = (float*)carve((size_t)NB * 16384 * 4);
  float*    norms  = (float*)carve(NB * 4);
  float*    vhid   = (float*)carve((size_t)NB * 256 * 4);

  // ---- stage fp16 copies of x and weights ----
  {
    int n = NB * 64 * 900;
    cvt_f16_kernel<<<(n + 255) / 256, 256, 0, stream>>>(x, x16, n);
    for (int l = 0; l < 6; ++l)
      cvt_f16_kernel<<<(wcnt[l] + 255) / 256, 256, 0, stream>>>(conv_w[l], w16[l], wcnt[l]);
    for (int l = 0; l < 6; ++l)
      cvt_f16_kernel<<<(gwcnt[l] + 255) / 256, 256, 0, stream>>>(gwsrc[l], gw16[l], gwcnt[l]);
  }

  // ---- conv pyramid: conv(WMMA) -> GN stats -> GN+ReLU(+f16) ----
  const _Float16* cur = x16;
  for (int l = 0; l < 6; ++l) {
    const int Cin = cs[l][0], Hin = cs[l][1], Win = cs[l][2], Cout = cs[l][3];
    const int Ks = cs[l][4], st = cs[l][5], pd = cs[l][6], Ho = cs[l][7], Wo = cs[l][8];
    const int groups = cs[l][9];
    const int HW = Ho * Wo;
    dim3 cgrid(HW / 16, Cout / 16, NB);
    conv_wmma_kernel<<<cgrid, 32, 0, stream>>>(cur, w16[l], c32,
                                               Cin, Hin, Win, Cout, Ks, st, pd, Ho, Wo);
    gn_stats_kernel<<<NB * groups, 1024, 0, stream>>>(c32, stats, Cout, HW, groups);
    _Float16* o16 = (l & 1) ? h16b : h16a;
    float* o32 = (l == 5) ? g32l : nullptr;
    size_t total = (size_t)NB * Cout * HW;
    gn_norm_kernel<<<(unsigned)((total + 255) / 256), 256, 0, stream>>>(
        c32, stats, gnw[l], gnb[l], o16, o32, Cout, HW, groups);
    cur = o16;
  }

  // ---- pool to node features + scores + top-k(500) mask ----
  pool_kernel<<<dim3(NN, NB), 128, 0, stream>>>(g32l, pf, pf16);
  row_norm_kernel<<<NB * NN, 256, 0, stream>>>(pf, s1, 128);
  topk_kernel<<<NB, 512, 0, stream>>>(s1, nullptr, NN, TK1, maskb, nullptr);

  // ---- GAT layer 1 (K=128) ----
  {
    dim3 ggrid(NB * NN / 16, 16);
    gemm_lin_kernel<<<ggrid, 32, 0, stream>>>(pf16, gw16[0], g1_lb, xl, 128);
    gemm_lin_kernel<<<ggrid, 32, 0, stream>>>(pf16, gw16[1], g1_rb, xr, 128);
    gemm_lin_kernel<<<ggrid, 32, 0, stream>>>(pf16, gw16[2], g1_bi, res, 128);
    gat_agg_kernel<<<dim3(NN, NB), 256, 0, stream>>>(xl, xr, g1_at, maskb, agg);
    int n = NB * NN * 256;
    gat_combine_kernel<<<(n + 255) / 256, 256, 0, stream>>>(agg, res, h1, h1_16, n);
  }
  // ---- GAT layer 2 (K=256) ----
  {
    dim3 ggrid(NB * NN / 16, 16);
    gemm_lin_kernel<<<ggrid, 32, 0, stream>>>(h1_16, gw16[3], g2_lb, xl, 256);
    gemm_lin_kernel<<<ggrid, 32, 0, stream>>>(h1_16, gw16[4], g2_rb, xr, 256);
    gemm_lin_kernel<<<ggrid, 32, 0, stream>>>(h1_16, gw16[5], g2_bi, res, 256);
    gat_agg_kernel<<<dim3(NN, NB), 256, 0, stream>>>(xl, xr, g2_at, maskb, agg);
    int n = NB * NN * 256;
    gat_combine_kernel<<<(n + 255) / 256, 256, 0, stream>>>(agg, res, h2, (_Float16*)nullptr, n);
  }

  // ---- top-k(300) among the 500 selected, gather + row-normalize ----
  row_norm_kernel<<<NB * NN, 256, 0, stream>>>(h2, s2f, 256);
  topk_kernel<<<NB, 512, 0, stream>>>(s2f, maskb, NN, TK2, nullptr, sel);
  gather_norm_kernel<<<dim3(TK2, NB), 256, 0, stream>>>(h2, sel, gsel);

  // ---- NetVLAD ----
  nv_act_kernel<<<dim3(TK2, NB), 64, 0, stream>>>(gsel, nv_cw, nv_cb, actn);
  nv_vlad_kernel<<<dim3(64, NB), 256, 0, stream>>>(actn, gsel, nv_cw2, vlad);
  nv_intra_kernel<<<dim3(64, NB), 256, 0, stream>>>(vlad);
  vec_norm_kernel<<<NB, 256, 0, stream>>>(vlad, norms, 16384);
  zero_kernel<<<(NB * 256 + 255) / 256, 256, 0, stream>>>(vhid, NB * 256);
  nv_hidden_kernel<<<dim3(64, NB), 256, 0, stream>>>(vlad, norms, nv_hw, vhid);
  nv_final_kernel<<<1, 256, 0, stream>>>(vhid, nv_bg, nv_bb, nv_gw, nv_gbp, outp);
}
